// AttentionBlock_17575006175896
// MI455X (gfx1250) — matrix operand order repaired and checked by reading, not compiled, not run on hardware
//
#include <hip/hip_runtime.h>

typedef __attribute__((ext_vector_type(16))) _Float16 v16h;
typedef __attribute__((ext_vector_type(8)))  float    v8f;
typedef __attribute__((ext_vector_type(8)))  _Float16 h8;
typedef __attribute__((ext_vector_type(4)))  _Float16 h4;

constexpr int CC    = 64;                 // channels
constexpr int NTOT  = 64 * 64 * 64;       // 262144 spatial positions
constexpr int TN    = 256;                // n-columns per tile
constexpr int LDX   = TN + 8;             // LDS ld (halfs) for [chan][pos] 64 x TN tiles
constexpr int LDC   = CC + 8;             // LDS ld (halfs) for [pos][chan] transposed tiles
constexpr int LDW   = 72;                 // LDS ld for 64x64 matrices
constexpr int BLK   = 256;                // threads per block = 8 waves (wave32)
constexpr int ITERS = 4;                  // n-tiles accumulated per block in kernel 1
constexpr int PBLK  = NTOT / (TN * ITERS);// 256 partial-S blocks per batch

// ---------------- WMMA fragment helpers (CDNA5 wave32 layouts, CDNA5 ISA) ----------------

__device__ __forceinline__ v8f wmma_f16(v16h a, v16h b, v8f c) {
  return __builtin_amdgcn_wmma_f32_16x16x32_f16(false, a, false, b, (short)0, c, false, false);
}

// A-matrix 16x32 f16 from row-major LDS [M][K], top-left (row0,k0).
// Lane m=lane&15, g=lane>>4.  half h -> K = k0 + 8*g + (h&7) + 16*(h>>3).  (2x b128 per lane)
__device__ __forceinline__ v16h load_A(const _Float16* lds, int ld, int row0, int k0) {
  const int lane = threadIdx.x & 31;
  const int g = lane >> 4, m = lane & 15;
  const _Float16* p = lds + (row0 + m) * ld + k0 + 8 * g;
  v16h a;
#pragma unroll
  for (int h = 0; h < 16; ++h) a[h] = p[(h & 7) + ((h >> 3) << 4)];
  return a;
}

// B-matrix 32x16 f16 where LDS holds element (k,n) at lds[(n0+n)*ld + k].
// Per-lane half h -> K = k0 + 8*g + (h&7) + 16*(h>>3) (matches load_A) -> 2x b128.
__device__ __forceinline__ v16h load_Bt(const _Float16* lds, int ld, int k0, int n0) {
  const int lane = threadIdx.x & 31;
  const int g = lane >> 4, n = lane & 15;
  const _Float16* p = lds + (n0 + n) * ld + k0 + 8 * g;
  v16h b;
#pragma unroll
  for (int h = 0; h < 16; ++h) b[h] = p[(h & 7) + ((h >> 3) << 4)];
  return b;
}

// D (v8f) -> f16 LDS row-major [row][col], +bias.  Row = row0+r+8g, col = n0+(lane&15).
__device__ __forceinline__ void store_D_rm(_Float16* lds, int ld, int row0, int n0,
                                           v8f c, const float* bias) {
  const int lane = threadIdx.x & 31;
  const int g = lane >> 4, n = lane & 15;
#pragma unroll
  for (int r = 0; r < 8; ++r) {
    const int row = row0 + r + 8 * g;
    lds[row * ld + n0 + n] = (_Float16)(c[r] + bias[row]);
  }
}

// D (v8f) -> f16 LDS column-major [col][row], +bias. One aligned 16B store per lane.
__device__ __forceinline__ void store_D_cm(_Float16* lds, int ld, int row0, int n0,
                                           v8f c, const float* bias) {
  const int lane = threadIdx.x & 31;
  const int g = lane >> 4, n = lane & 15;
  h8 hv;
#pragma unroll
  for (int r = 0; r < 8; ++r) hv[r] = (_Float16)(c[r] + bias[row0 + r + 8 * g]);
  *reinterpret_cast<h8*>(&lds[(n0 + n) * ld + row0 + 8 * g]) = hv;
}

// Stage: global f32 [chan][pos] tile -> LDS f16 transposed [pos][chan]
__device__ __forceinline__ void stage_x_T(const float* __restrict__ x, size_t base,
                                          _Float16* sXt) {
  for (int i = threadIdx.x; i < CC * (TN / 4); i += BLK) {
    const int row = i >> 6, c4 = i & 63;  // 64 float4 per channel row
    const float4 v = *reinterpret_cast<const float4*>(x + base + (size_t)row * NTOT + c4 * 4);
    const int n = c4 * 4;
    sXt[(n + 0) * LDC + row] = (_Float16)v.x;
    sXt[(n + 1) * LDC + row] = (_Float16)v.y;
    sXt[(n + 2) * LDC + row] = (_Float16)v.z;
    sXt[(n + 3) * LDC + row] = (_Float16)v.w;
  }
}

// ---------------- Kernel 1: Q/K projection + partial S = Q K^T -----------------------------

__global__ void __launch_bounds__(BLK)
qk_attn_partial_kernel(const float* __restrict__ x,
                       const float* __restrict__ wq, const float* __restrict__ bq,
                       const float* __restrict__ wk, const float* __restrict__ bk,
                       float* __restrict__ part) {
  __shared__ _Float16 sWq[CC * LDW];
  __shared__ _Float16 sWk[CC * LDW];
  __shared__ float    sBq[CC];
  __shared__ float    sBk[CC];
  __shared__ _Float16 sXt[TN * LDC];   // x tile, transposed [pos][chan]
  __shared__ _Float16 sQ [CC * LDX];   // q tile, [chan][pos]
  __shared__ _Float16 sK [CC * LDX];   // k tile, [chan][pos]

  const int tid  = threadIdx.x;
  const int wave = tid >> 5;
  const int lane = tid & 31;
  const int g = lane >> 4, ln = lane & 15;
  const int b    = blockIdx.y;
  const int pblk = blockIdx.x;

  for (int i = tid; i < CC * CC; i += BLK) {
    sWq[(i >> 6) * LDW + (i & 63)] = (_Float16)wq[i];
    sWk[(i >> 6) * LDW + (i & 63)] = (_Float16)wk[i];
  }
  if (tid < CC) { sBq[tid] = bq[tid]; sBk[tid] = bk[tid]; }

  // S-tile assignment: this wave owns (trS, tcS) and (trS+2, tcS)
  const int tcS = wave & 3, trS = wave >> 2;
  v8f accS[2] = {};

  for (int it = 0; it < ITERS; ++it) {
    const int n0 = (pblk + it * PBLK) * TN;
    __syncthreads();
    stage_x_T(x, (size_t)b * CC * NTOT + n0, sXt);
    if (it + 1 < ITERS) {  // prefetch next tile (global_prefetch)
      const size_t nb = (size_t)b * CC * NTOT + (size_t)(pblk + (it + 1) * PBLK) * TN;
      const int row = tid >> 2, seg = tid & 3;
      __builtin_prefetch((const void*)(x + nb + (size_t)row * NTOT + seg * 64), 0, 1);
    }
    __syncthreads();
    // ---- Q = Wq*X + bq, K = Wk*X + bk.  B fragments hoisted per position-column. ----
    for (int s = 0; s < 2; ++s) {
      const int tc = wave + 8 * s;                 // waves jointly cover tc = 0..15
      const v16h b0 = load_Bt(sXt, LDC, 0,  tc * 16);
      const v16h b1 = load_Bt(sXt, LDC, 32, tc * 16);
#pragma unroll
      for (int wsel = 0; wsel < 2; ++wsel) {
        const _Float16* W  = wsel ? sWk : sWq;
        const float*    Bs = wsel ? sBk : sBq;
        _Float16*       Dd = wsel ? sK  : sQ;
#pragma unroll
        for (int tr = 0; tr < 4; ++tr) {
          v8f acc = {};
          acc = wmma_f16(load_A(W, LDW, tr * 16, 0),  b0, acc);
          acc = wmma_f16(load_A(W, LDW, tr * 16, 32), b1, acc);
          store_D_rm(Dd, LDX, tr * 16, tc * 16, acc, Bs);
        }
      }
    }
    __syncthreads();
    // ---- S += Q * K^T over this tile's TN columns; B fragment shared by both accs ----
#pragma unroll
    for (int kk = 0; kk < TN / 32; ++kk) {
      const v16h bf = load_Bt(sK, LDX, kk * 32, tcS * 16);
      accS[0] = wmma_f16(load_A(sQ, LDX, trS * 16,       kk * 32), bf, accS[0]);
      accS[1] = wmma_f16(load_A(sQ, LDX, (trS + 2) * 16, kk * 32), bf, accS[1]);
    }
  }

  // ---- write this block's 64x64 partial S (deterministic; reduced in kernel 2) ----
  float* pout = part + ((size_t)b * PBLK + pblk) * (CC * CC);
  for (int j = 0; j < 2; ++j) {
    const int tr = trS + 2 * j;
#pragma unroll
    for (int r = 0; r < 8; ++r)
      pout[(tr * 16 + r + 8 * g) * CC + tcS * 16 + ln] = accS[j][r];
  }
}

// ---------------- Kernel 2: reduce partials + scale + row softmax --------------------------

__global__ void __launch_bounds__(64)
attn_softmax_kernel(const float* __restrict__ part, float* __restrict__ P) {
  __shared__ float red[CC];
  const int b = blockIdx.x >> 6, c = blockIdx.x & 63, d = threadIdx.x;
  const float* pp = part + (size_t)b * PBLK * (CC * CC) + c * CC + d;
  float s = 0.f;
  for (int p = 0; p < PBLK; ++p) s += pp[(size_t)p * (CC * CC)];
  s *= 0.125f;  // 1/sqrt(64)
  red[d] = s; __syncthreads();
  for (int o = 32; o > 0; o >>= 1) { if (d < o) red[d] = fmaxf(red[d], red[d + o]); __syncthreads(); }
  const float m = red[0]; __syncthreads();
  const float e = __expf(s - m);
  red[d] = e; __syncthreads();
  for (int o = 32; o > 0; o >>= 1) { if (d < o) red[d] += red[d + o]; __syncthreads(); }
  P[((size_t)b * CC + c) * CC + d] = e / red[0];
}

// ---------------- Kernel 3: V = Wv*x + bv ; out = P*V + x ----------------------------------

__global__ void __launch_bounds__(BLK)
attn_out_kernel(const float* __restrict__ x,
                const float* __restrict__ wv, const float* __restrict__ bv,
                const float* __restrict__ P,
                float* __restrict__ out) {
  __shared__ _Float16 sWv[CC * LDW];
  __shared__ _Float16 sP [CC * LDW];
  __shared__ float    sBv[CC];
  __shared__ _Float16 sXt[TN * LDC];   // x tile, transposed [pos][chan]
  __shared__ _Float16 sVt[TN * LDC];   // v tile, transposed [pos][chan]

  const int tid  = threadIdx.x;
  const int wave = tid >> 5;
  const int lane = tid & 31;
  const int g = lane >> 4, ln = lane & 15;
  const int b  = blockIdx.y;
  const int n0 = blockIdx.x * TN;

  for (int i = tid; i < CC * CC; i += BLK) {
    sWv[(i >> 6) * LDW + (i & 63)] = (_Float16)wv[i];
    sP [(i >> 6) * LDW + (i & 63)] = (_Float16)P[(size_t)b * CC * CC + i];
  }
  if (tid < CC) sBv[tid] = bv[tid];

  stage_x_T(x, (size_t)b * CC * NTOT + n0, sXt);
  __syncthreads();

  // ---- V = Wv*X + bv, stored transposed; B fragments hoisted ----
  for (int s = 0; s < 2; ++s) {
    const int tc = wave + 8 * s;
    const v16h b0 = load_Bt(sXt, LDC, 0,  tc * 16);
    const v16h b1 = load_Bt(sXt, LDC, 32, tc * 16);
#pragma unroll
    for (int tr = 0; tr < 4; ++tr) {
      v8f acc = {};
      acc = wmma_f16(load_A(sWv, LDW, tr * 16, 0),  b0, acc);
      acc = wmma_f16(load_A(sWv, LDW, tr * 16, 32), b1, acc);
      store_D_cm(sVt, LDC, tr * 16, tc * 16, acc, sBv);
    }
  }
  __syncthreads();

  // ---- O = P * V ; residual add with fresh f32 x (L2-resident), store ----
  for (int s = 0; s < 2; ++s) {
    const int tc = wave + 8 * s;
    const v16h b0 = load_Bt(sVt, LDC, 0,  tc * 16);
    const v16h b1 = load_Bt(sVt, LDC, 32, tc * 16);
#pragma unroll
    for (int tr = 0; tr < 4; ++tr) {
      v8f acc = {};
      acc = wmma_f16(load_A(sP, LDW, tr * 16, 0),  b0, acc);
      acc = wmma_f16(load_A(sP, LDW, tr * 16, 32), b1, acc);
#pragma unroll
      for (int r = 0; r < 8; ++r) {
        const int row = tr * 16 + r + 8 * g;
        const size_t off = (size_t)b * CC * NTOT + (size_t)row * NTOT + n0 + tc * 16 + ln;
        out[off] = acc[r] + x[off];
      }
    }
  }
}

// ---------------- launch ----------------

extern "C" void kernel_launch(void* const* d_in, const int* in_sizes, int n_in,
                              void* d_out, int out_size, void* d_ws, size_t ws_size,
                              hipStream_t stream) {
  (void)in_sizes; (void)n_in; (void)out_size; (void)ws_size;
  const float* x  = (const float*)d_in[0];
  const float* wq = (const float*)d_in[1];
  const float* bq = (const float*)d_in[2];
  const float* wk = (const float*)d_in[3];
  const float* bk = (const float*)d_in[4];
  const float* wv = (const float*)d_in[5];
  const float* bv = (const float*)d_in[6];
  float* out  = (float*)d_out;
  float* part = (float*)d_ws;                          // 2*PBLK*64*64 floats = 8 MB
  float* P    = part + (size_t)2 * PBLK * CC * CC;     // 2*64*64 floats

  qk_attn_partial_kernel<<<dim3(PBLK, 2), BLK, 0, stream>>>(x, wq, bq, wk, bk, part);
  attn_softmax_kernel<<<dim3(2 * CC), CC, 0, stream>>>(part, P);
  attn_out_kernel<<<dim3(NTOT / TN, 2), BLK, 0, stream>>>(x, wv, bv, P, out);
}
